// Block_48473000902970
// MI455X (gfx1250) — matrix operand order repaired and checked by reading, not compiled, not run on hardware
//
#include <hip/hip_runtime.h>
#include <hip/hip_bf16.h>

// ---------------------------------------------------------------------------
// Types for CDNA5 WMMA
// ---------------------------------------------------------------------------
typedef __attribute__((ext_vector_type(16))) __bf16    v16bf;
typedef __attribute__((ext_vector_type(8)))  float     v8f;
typedef __attribute__((ext_vector_type(4)))  unsigned  u32x4;

union AFrag {
    v16bf v;
    u32x4 q[2];
};

__device__ inline v8f wmma_bf16(const AFrag& a, const AFrag& b, v8f c) {
    // D = A(16x32 bf16) * B(32x16 bf16) + C(16x16 f32)
    return __builtin_amdgcn_wmma_f32_16x16x32_bf16(
        /*neg_a=*/false, a.v, /*neg_b=*/false, b.v,
        /*c_mod=*/(short)0, c, /*reuse_a=*/false, /*reuse_b=*/false);
}

// CDNA5 async global->LDS copy (16B per lane), tracked by ASYNCcnt.
__device__ inline void async_copy16(unsigned lds_addr, const void* gptr) {
    asm volatile("global_load_async_to_lds_b128 %0, %1, off"
                 :: "v"(lds_addr), "v"((unsigned long long)gptr)
                 : "memory");
}
__device__ inline void wait_async0() {
    asm volatile("s_wait_asynccnt 0" ::: "memory");
}

__device__ inline unsigned short f2bf(float f) {
    unsigned u = __float_as_uint(f);
    unsigned r = (u + 0x7FFFu + ((u >> 16) & 1u)) >> 16;
    return (unsigned short)r;
}

__device__ inline float rowmax16(float v) {
#pragma unroll
    for (int off = 8; off > 0; off >>= 1)
        v = fmaxf(v, __shfl_xor(v, off, 16));
    return v;
}
__device__ inline float rowsum16(float v) {
#pragma unroll
    for (int off = 8; off > 0; off >>= 1)
        v += __shfl_xor(v, off, 16);
    return v;
}

// ---------------------------------------------------------------------------
// Problem constants
// ---------------------------------------------------------------------------
#define BB   128
#define TT   256
#define CC   384
#define HH   6
#define DD   64
#define BT   (BB * TT)          // 32768
#define HD   (HH * DD)          // 384
#define C4   (4 * CC)           // 1536
#define QKVN (3 * HD)           // 1152

// ---------------------------------------------------------------------------
// Weight conversion / packing
// ---------------------------------------------------------------------------
__global__ void f2bf_kernel(const float* __restrict__ in,
                            unsigned short* __restrict__ out, int n) {
    int i = blockIdx.x * blockDim.x + threadIdx.x;
    if (i < n) out[i] = f2bf(in[i]);
}

// Pack Wq/Wk/Wv (H,C,D) fp32 -> bf16 [C][3*H*D] row-major: col = sel*384 + h*64 + d
__global__ void pack_qkv_kernel(const float* __restrict__ Wq,
                                const float* __restrict__ Wk,
                                const float* __restrict__ Wv,
                                unsigned short* __restrict__ out) {
    int idx = blockIdx.x * blockDim.x + threadIdx.x;   // 0 .. 384*1152-1 (exact)
    int k   = idx / QKVN;
    int n   = idx - k * QKVN;
    int sel = n / HD;
    int hd  = n - sel * HD;
    int h   = hd >> 6;
    int d   = hd & 63;
    const float* W = (sel == 0) ? Wq : (sel == 1) ? Wk : Wv;
    out[idx] = f2bf(W[((size_t)h * CC + k) * DD + d]);
}

// ---------------------------------------------------------------------------
// LayerNorm over C=384, output bf16.  block=128, grid=BT.
// ---------------------------------------------------------------------------
__global__ __launch_bounds__(128) void ln_kernel(const float* __restrict__ x,
                                                 const float* __restrict__ g,
                                                 const float* __restrict__ be,
                                                 unsigned short* __restrict__ out) {
    __shared__ float red[128];
    const int row = blockIdx.x, tid = threadIdx.x;
    const float* xr = x + (size_t)row * CC;
    float v0 = xr[tid], v1 = xr[tid + 128], v2 = xr[tid + 256];
    red[tid] = v0 + v1 + v2;
    __syncthreads();
#pragma unroll
    for (int st = 64; st > 0; st >>= 1) {
        if (tid < st) red[tid] += red[tid + st];
        __syncthreads();
    }
    float mean = red[0] * (1.0f / CC);
    __syncthreads();
    float d0 = v0 - mean, d1 = v1 - mean, d2 = v2 - mean;
    red[tid] = d0 * d0 + d1 * d1 + d2 * d2;
    __syncthreads();
#pragma unroll
    for (int st = 64; st > 0; st >>= 1) {
        if (tid < st) red[tid] += red[tid + st];
        __syncthreads();
    }
    float rstd = rsqrtf(red[0] * (1.0f / CC) + 1e-5f);
    unsigned short* o = out + (size_t)row * CC;
    o[tid]       = f2bf(d0 * rstd * g[tid]       + be[tid]);
    o[tid + 128] = f2bf(d1 * rstd * g[tid + 128] + be[tid + 128]);
    o[tid + 256] = f2bf(d2 * rstd * g[tid + 256] + be[tid + 256]);
}

// ---------------------------------------------------------------------------
// Generic bf16 WMMA GEMM: C[M,N] = A[M,K] * B[K,N] (+bias, relu, residual)
// Block tile 128x64, 8 waves, each wave does a 16x64 strip. K step = 64
// (two WMMA K-blocks per staging round).  A tile staged via async global->LDS.
// grid = (N/64, M/128), block = 256.
// ---------------------------------------------------------------------------
__global__ __launch_bounds__(256) void gemm_bf16(
    const unsigned short* __restrict__ A, int lda,
    const unsigned short* __restrict__ B, int ldb,
    int K,
    const float* __restrict__ bias,
    const float* __restrict__ residual, int ldr,
    float* __restrict__ outF,
    unsigned short* __restrict__ outB, int ldo,
    int do_relu) {
    __shared__ __align__(16) unsigned short As[128 * 64];  // [m][k]   16 KB
    __shared__ __align__(16) unsigned short Bs[64 * 64];   // [n][k]    8 KB

    const int tid  = threadIdx.x;
    const int wave = tid >> 5;
    const int lane = tid & 31;
    const int bm   = blockIdx.y * 128;
    const int bn   = blockIdx.x * 64;

    v8f acc[4];
#pragma unroll
    for (int t = 0; t < 4; ++t) acc[t] = v8f{};

    const int arow = lane & 15;
    const int kb   = (lane >> 4) * 8;    // A grouped-K base
    const int kb2  = (lane >> 4) * 16;   // contiguous-K base (unused; B uses kb)

    for (int k0 = 0; k0 < K; k0 += 64) {
        __syncthreads();
        // stage A: 128x64 bf16 via async global->LDS (4 x 16B per thread)
#pragma unroll
        for (int p = 0; p < 4; ++p) {
            int idx = tid + p * 256;
            int r = idx >> 3, c8 = (idx & 7) * 8;
            async_copy16((unsigned)(size_t)&As[r * 64 + c8],
                         &A[(size_t)(bm + r) * lda + k0 + c8]);
        }
        // stage B transposed: 64(k) x 64(n) -> Bs[n][k]
#pragma unroll
        for (int p = 0; p < 2; ++p) {
            int idx = tid + p * 256;
            int k = idx >> 3, n8 = (idx & 7) * 8;
            u32x4 bv = *(const u32x4*)&B[(size_t)(k0 + k) * ldb + bn + n8];
            const unsigned short* bu = (const unsigned short*)&bv;
#pragma unroll
            for (int j = 0; j < 8; ++j) Bs[(n8 + j) * 64 + k] = bu[j];
        }
        wait_async0();
        __syncthreads();

#pragma unroll
        for (int ks = 0; ks < 2; ++ks) {
            AFrag af;
            const int abase = (wave * 16 + arow) * 64 + ks * 32;
            af.q[0] = *(const u32x4*)&As[abase + kb];
            af.q[1] = *(const u32x4*)&As[abase + kb + 16];
            AFrag bf[4];
#pragma unroll
            for (int t = 0; t < 4; ++t) {
                const int bbase = (t * 16 + arow) * 64 + ks * 32;
                bf[t].q[0] = *(const u32x4*)&Bs[bbase + kb];
                bf[t].q[1] = *(const u32x4*)&Bs[bbase + kb + 16];
            }
#pragma unroll
            for (int t = 0; t < 4; ++t)
                acc[t] = wmma_bf16(af, bf[t], acc[t]);
        }
    }

    // epilogue
    const int nl = lane & 15, r8 = (lane >> 4) * 8;
#pragma unroll
    for (int t = 0; t < 4; ++t) {
        int col = bn + t * 16 + nl;
        float bv = bias ? bias[col] : 0.0f;
#pragma unroll
        for (int vr = 0; vr < 8; ++vr) {
            int row = bm + wave * 16 + r8 + vr;
            float v = acc[t][vr] + bv;
            if (do_relu) v = fmaxf(v, 0.0f);
            if (residual) v += residual[(size_t)row * ldr + col];
            if (outF) outF[(size_t)row * ldo + col] = v;
            if (outB) outB[(size_t)row * ldo + col] = f2bf(v);
        }
    }
}

// ---------------------------------------------------------------------------
// Fused causal flash attention.
// qkv: bf16 [BT][1152]  (Q | K | V, each [H][64] per token)
// out: bf16 [BT][384]   (heads concatenated)
// grid = (2, H, B), block = 256 (8 waves; wave w owns Q rows mt*128+w*16..+15)
// ---------------------------------------------------------------------------
__global__ __launch_bounds__(256) void attn_kernel(
    const unsigned short* __restrict__ qkv,
    unsigned short* __restrict__ outO) {
    __shared__ __align__(16) unsigned short Ks[32 * 64];      // [s][d]
    __shared__ __align__(16) unsigned short Vt[64 * 32];      // [d][s]
    __shared__ __align__(16) unsigned short Ps[8][16 * 32];   // per-wave P tile

    const int tid = threadIdx.x, wave = tid >> 5, lane = tid & 31;
    const int mt = blockIdx.x, h = blockIdx.y, b = blockIdx.z;
    const size_t browbase = (size_t)b * TT;
    const int r0 = mt * 128 + wave * 16;                 // wave's first query row
    const int koff = HD + h * 64, voff = 2 * HD + h * 64;
    const int kb  = (lane >> 4) * 8;
    const int kb2 = (lane >> 4) * 16;
    const int nl  = lane & 15;

    // Q fragments (A operand, 16 rows x 64 d, two K-steps of 32)
    AFrag qf[2];
    {
        const unsigned short* qp = qkv + (browbase + r0 + nl) * QKVN + h * 64;
#pragma unroll
        for (int ks = 0; ks < 2; ++ks) {
            qf[ks].q[0] = *(const u32x4*)(qp + ks * 32 + kb);
            qf[ks].q[1] = *(const u32x4*)(qp + ks * 32 + kb + 16);
        }
    }

    float m[8], l[8];
#pragma unroll
    for (int i = 0; i < 8; ++i) { m[i] = -__builtin_inff(); l[i] = 0.0f; }
    v8f o0{}, o1{}, o2{}, o3{};

    const int nchunks = mt * 4 + 4;                      // 32-key chunks (causal)
    const float scale = 0.051031036307982884f;           // 384^-0.5

    for (int j = 0; j < nchunks; ++j) {
        __syncthreads();
        // stage K chunk [32 s][64 d] (async) and V chunk transposed [64 d][32 s]
        {
            int srow = tid >> 3, d8 = (tid & 7) * 8;
            const unsigned short* kp = qkv + (browbase + j * 32 + srow) * QKVN;
            async_copy16((unsigned)(size_t)&Ks[srow * 64 + d8], kp + koff + d8);
            u32x4 vv = *(const u32x4*)(kp + voff + d8);
            const unsigned short* vu = (const unsigned short*)&vv;
#pragma unroll
            for (int i = 0; i < 8; ++i) Vt[(d8 + i) * 32 + srow] = vu[i];
        }
        wait_async0();
        __syncthreads();

        if (j * 32 <= r0 + 15) {                         // wave-uniform
            // ---- scores: S[16 x 32] = Q x K^T ----
            v8f s0{}, s1{};
#pragma unroll
            for (int ks = 0; ks < 2; ++ks) {
                AFrag kf0, kf1;
                kf0.q[0] = *(const u32x4*)&Ks[nl * 64 + ks * 32 + kb];
                kf0.q[1] = *(const u32x4*)&Ks[nl * 64 + ks * 32 + kb + 16];
                kf1.q[0] = *(const u32x4*)&Ks[(16 + nl) * 64 + ks * 32 + kb];
                kf1.q[1] = *(const u32x4*)&Ks[(16 + nl) * 64 + ks * 32 + kb + 16];
                s0 = wmma_bf16(qf[ks], kf0, s0);
                s1 = wmma_bf16(qf[ks], kf1, s1);
            }
            // ---- online softmax ----
            const int rowbase = r0 + (lane >> 4) * 8;
            const int sc0 = j * 32 + nl, sc1 = sc0 + 16;
            float al[8];
#pragma unroll
            for (int vr = 0; vr < 8; ++vr) {
                int row = rowbase + vr;
                float a  = (sc0 <= row) ? s0[vr] * scale : -__builtin_inff();
                float bq = (sc1 <= row) ? s1[vr] * scale : -__builtin_inff();
                float v = rowmax16(fmaxf(a, bq));
                float mn = fmaxf(m[vr], v);
                float alpha = __expf(m[vr] - mn);
                m[vr] = mn;
                float p0 = __expf(a - mn), p1 = __expf(bq - mn);
                l[vr] = l[vr] * alpha + rowsum16(p0 + p1);
                al[vr] = alpha;
                int rl = (lane >> 4) * 8 + vr;
                Ps[wave][rl * 32 + nl]      = f2bf(p0);
                Ps[wave][rl * 32 + 16 + nl] = f2bf(p1);
            }
#pragma unroll
            for (int vr = 0; vr < 8; ++vr) {
                o0[vr] *= al[vr]; o1[vr] *= al[vr];
                o2[vr] *= al[vr]; o3[vr] *= al[vr];
            }
            // intra-wave LDS RAW: drain DS counter before fragment gather
            asm volatile("s_wait_dscnt 0" ::: "memory");
            AFrag pf;
            pf.q[0] = *(const u32x4*)&Ps[wave][nl * 32 + kb];
            pf.q[1] = *(const u32x4*)&Ps[wave][nl * 32 + kb + 16];
            // ---- O += P x V ----
            {
                AFrag vf0, vf1, vf2, vf3;
                vf0.q[0] = *(const u32x4*)&Vt[(0 * 16 + nl) * 32 + kb];
                vf0.q[1] = *(const u32x4*)&Vt[(0 * 16 + nl) * 32 + kb + 16];
                vf1.q[0] = *(const u32x4*)&Vt[(1 * 16 + nl) * 32 + kb];
                vf1.q[1] = *(const u32x4*)&Vt[(1 * 16 + nl) * 32 + kb + 16];
                vf2.q[0] = *(const u32x4*)&Vt[(2 * 16 + nl) * 32 + kb];
                vf2.q[1] = *(const u32x4*)&Vt[(2 * 16 + nl) * 32 + kb + 16];
                vf3.q[0] = *(const u32x4*)&Vt[(3 * 16 + nl) * 32 + kb];
                vf3.q[1] = *(const u32x4*)&Vt[(3 * 16 + nl) * 32 + kb + 16];
                o0 = wmma_bf16(pf, vf0, o0);
                o1 = wmma_bf16(pf, vf1, o1);
                o2 = wmma_bf16(pf, vf2, o2);
                o3 = wmma_bf16(pf, vf3, o3);
            }
        }
    }

    // normalize + write bf16 [BT][384]
    const int r8 = (lane >> 4) * 8;
#pragma unroll
    for (int vr = 0; vr < 8; ++vr) {
        float inv = 1.0f / l[vr];
        unsigned short* op = outO + (browbase + r0 + r8 + vr) * HD + h * 64;
        op[0 * 16 + nl] = f2bf(o0[vr] * inv);
        op[1 * 16 + nl] = f2bf(o1[vr] * inv);
        op[2 * 16 + nl] = f2bf(o2[vr] * inv);
        op[3 * 16 + nl] = f2bf(o3[vr] * inv);
    }
}

// ---------------------------------------------------------------------------
// Host launcher
// ---------------------------------------------------------------------------
extern "C" void kernel_launch(void* const* d_in, const int* in_sizes, int n_in,
                              void* d_out, int out_size, void* d_ws, size_t ws_size,
                              hipStream_t stream) {
    const float* x   = (const float*)d_in[0];
    const float* Wq  = (const float*)d_in[1];
    const float* Wk  = (const float*)d_in[2];
    const float* Wv  = (const float*)d_in[3];
    const float* Wo  = (const float*)d_in[4];
    const float* bo  = (const float*)d_in[5];
    const float* W1  = (const float*)d_in[6];
    const float* b1  = (const float*)d_in[7];
    const float* W2  = (const float*)d_in[8];
    const float* b2  = (const float*)d_in[9];
    const float* g1  = (const float*)d_in[10];
    const float* be1 = (const float*)d_in[11];
    const float* g2  = (const float*)d_in[12];
    const float* be2 = (const float*)d_in[13];
    float* out = (float*)d_out;

    // workspace layout (all region sizes are multiples of 256 bytes)
    char* ws = (char*)d_ws;
    size_t off = 0;
    unsigned short* xn1  = (unsigned short*)(ws + off); off += (size_t)BT * CC * 2;    // 25.2 MB
    unsigned short* qkvW = (unsigned short*)(ws + off); off += (size_t)CC * QKVN * 2;  // 0.9 MB
    unsigned short* WoB  = (unsigned short*)(ws + off); off += (size_t)HD * CC * 2;    // 0.3 MB
    unsigned short* W1B  = (unsigned short*)(ws + off); off += (size_t)CC * C4 * 2;    // 1.2 MB
    unsigned short* W2B  = (unsigned short*)(ws + off); off += (size_t)C4 * CC * 2;    // 1.2 MB
    unsigned short* qkvb = (unsigned short*)(ws + off); off += (size_t)BT * QKVN * 2;  // 75.5 MB
    unsigned short* aO   = (unsigned short*)(ws + off); off += (size_t)BT * HD * 2;    // 25.2 MB
    float*          x2   = (float*)(ws + off);          off += (size_t)BT * CC * 4;    // 50.3 MB
    unsigned short* xn2  = (unsigned short*)(ws + off); off += (size_t)BT * CC * 2;    // 25.2 MB
    unsigned short* hbuf = (unsigned short*)(ws + off); off += (size_t)BT * C4 * 2;    // 100.7 MB

    // weight prep (bf16)
    pack_qkv_kernel<<<(CC * QKVN) / 256, 256, 0, stream>>>(Wq, Wk, Wv, qkvW);
    f2bf_kernel<<<(HD * CC + 255) / 256, 256, 0, stream>>>(Wo, WoB, HD * CC);
    f2bf_kernel<<<(CC * C4 + 255) / 256, 256, 0, stream>>>(W1, W1B, CC * C4);
    f2bf_kernel<<<(C4 * CC + 255) / 256, 256, 0, stream>>>(W2, W2B, C4 * CC);

    // xn1 = LN(x; g1, be1)
    ln_kernel<<<BT, 128, 0, stream>>>(x, g1, be1, xn1);

    // qkv = xn1 @ [Wq|Wk|Wv]
    gemm_bf16<<<dim3(QKVN / 64, BT / 128), 256, 0, stream>>>(
        xn1, CC, qkvW, QKVN, CC,
        nullptr, nullptr, 0, nullptr, qkvb, QKVN, 0);

    // fused causal attention -> aO (bf16, heads concatenated)
    attn_kernel<<<dim3(2, HH, BB), 256, 0, stream>>>(qkvb, aO);

    // x2 = x + aO @ Wo + bo
    gemm_bf16<<<dim3(CC / 64, BT / 128), 256, 0, stream>>>(
        aO, HD, WoB, CC, HD,
        bo, x, CC, x2, nullptr, CC, 0);

    // xn2 = LN(x2; g2, be2)
    ln_kernel<<<BT, 128, 0, stream>>>(x2, g2, be2, xn2);

    // h = relu(xn2 @ W1 + b1)
    gemm_bf16<<<dim3(C4 / 64, BT / 128), 256, 0, stream>>>(
        xn2, CC, W1B, C4, CC,
        b1, nullptr, 0, nullptr, hbuf, C4, 1);

    // out = x2 + h @ W2 + b2
    gemm_bf16<<<dim3(CC / 64, BT / 128), 256, 0, stream>>>(
        hbuf, C4, W2B, CC, C4,
        b2, x2, CC, out, nullptr, CC, 0);
}
